// MultimodalGatedDeltaNet_57698590655106
// MI455X (gfx1250) — matrix operand order repaired and checked by reading, not compiled, not run on hardware
//
#include <hip/hip_runtime.h>
#include <hip/hip_bf16.h>
#include <math.h>

#define B_ 2
#define L_ 1024
#define HS_ 2048
#define NH_ 8
#define HD_ 256
#define DV_ 512
#define KD_ 2048
#define VD_ 4096
#define E_ 3
#define ML_ (B_ * L_)          // 2048
#define SCALE_ 0.0625f         // HD^-0.5

typedef __attribute__((ext_vector_type(16))) __bf16 v16bf;
typedef __attribute__((ext_vector_type(8)))  float  v8f;
typedef __bf16 bf16;

// ---------------------------------------------------------------- helpers ---
__device__ __forceinline__ unsigned short f2bf(float f) {
  union { float f; unsigned u; } c; c.f = f;
  unsigned u = c.u;
  u += 0x7FFFu + ((u >> 16) & 1u);   // round-to-nearest-even
  return (unsigned short)(u >> 16);
}

__device__ __forceinline__ float sigmoidf_(float x) { return 1.f / (1.f + expf(-x)); }

// gfx1250 async global->LDS staging (guarded; falls back to plain copy)
#if defined(__HIP_DEVICE_COMPILE__) && defined(__gfx1250__)
#if __has_builtin(__builtin_amdgcn_global_load_async_to_lds_b32) && \
    __has_builtin(__builtin_amdgcn_s_wait_asynccnt)
#define HAVE_ASYNC_LDS 1
#endif
#endif

__device__ __forceinline__ void stage_f32(const float* g, float* l) {
#ifdef HAVE_ASYNC_LDS
  void* gp = (void*)g;   // drop const, generic AS
  void* lp = (void*)l;
  __builtin_amdgcn_global_load_async_to_lds_b32(
      (__attribute__((address_space(1))) int*)gp,
      (__attribute__((address_space(3))) int*)lp, 0, 0);
#else
  *l = *g;
#endif
}
__device__ __forceinline__ void stage_wait() {
#ifdef HAVE_ASYNC_LDS
  __builtin_amdgcn_s_wait_asynccnt(0);
#endif
}

// -------------------------------------------------------- fp32 -> bf16 ------
__global__ __launch_bounds__(256) void cvt_bf16_kernel(
    const float* __restrict__ in, bf16* __restrict__ out, long n) {
  long i = ((long)blockIdx.x * 256 + threadIdx.x) * 4;
  if (i + 3 >= n) return;
  float4 v = *(const float4*)(in + i);
  union { unsigned short u[4]; uint2 d; } r;
  r.u[0] = f2bf(v.x); r.u[1] = f2bf(v.y); r.u[2] = f2bf(v.z); r.u[3] = f2bf(v.w);
  *(uint2*)(out + i) = r.d;
}

// ------------------------------------------------------------- WMMA GEMM ----
// C[M,N] = A[M,K] * B[N,K]^T   (A,B bf16 K-major; C fp32).
// Batched via blockIdx.z = e*nh + h with separate strides.
// Block: 256 threads = 8 waves, 4(M) x 2(N); wave tile 16x64 (4 wmma / step).
// Out-of-range rows/cols are clamped to 0 (loads stay in-bounds); their
// garbage results are simply never stored, so the hot loop is branch-free.
union AFrag { uint4 q[2]; v16bf v; };

__global__ __launch_bounds__(256) void wmma_gemm_kernel(
    const bf16* __restrict__ A, const bf16* __restrict__ Bm,
    float* __restrict__ C, int M, int N, int K,
    int lda, int ldb, int ldc,
    long aSE, long aSH, long bSE, long bSH, long cSE, long cSH, int nh) {
  int z = blockIdx.z;
  int e = z / nh, h = z % nh;
  A  += (size_t)e * aSE + (size_t)h * aSH;
  Bm += (size_t)e * bSE + (size_t)h * bSH;
  C  += (size_t)e * cSE + (size_t)h * cSH;

  const int wid  = threadIdx.x >> 5;
  const int lane = threadIdx.x & 31;
  const int half = lane >> 4;       // which K half-group this lane holds
  const int r    = lane & 15;

  const int m0 = blockIdx.y * 64 + (wid >> 1) * 16;
  const int n0 = blockIdx.x * 128 + (wid & 1) * 64;

  const int arow = m0 + r;
  const int arc  = (arow < M) ? arow : 0;
  int bcol[4], bcc[4];
#pragma unroll
  for (int t = 0; t < 4; ++t) {
    bcol[t] = n0 + t * 16 + r;
    bcc[t]  = (bcol[t] < N) ? bcol[t] : 0;
  }

  v8f acc0 = {0.f, 0.f, 0.f, 0.f, 0.f, 0.f, 0.f, 0.f};
  v8f acc1 = acc0, acc2 = acc0, acc3 = acc0;

  // per-lane base pointers (advance by 32 elements per k-step)
  const bf16* ap  = A  + (size_t)arc * lda + half * 16;     // 16 contiguous
  const bf16* bp0 = Bm + (size_t)bcc[0] * ldb + half * 16;  // 16 contiguous
  const bf16* bp1 = Bm + (size_t)bcc[1] * ldb + half * 16;
  const bf16* bp2 = Bm + (size_t)bcc[2] * ldb + half * 16;
  const bf16* bp3 = Bm + (size_t)bcc[3] * ldb + half * 16;

#pragma unroll 2
  for (int k0 = 0; k0 < K; k0 += 32) {
    AFrag af, bf0, bf1, bf2, bf3;
    af.q[0]  = *(const uint4*)(ap);          // K base..base+7
    af.q[1]  = *(const uint4*)(ap + 8);      // K base+8..base+15
    bf0.q[0] = *(const uint4*)(bp0);
    bf0.q[1] = *(const uint4*)(bp0 + 8);
    bf1.q[0] = *(const uint4*)(bp1);
    bf1.q[1] = *(const uint4*)(bp1 + 8);
    bf2.q[0] = *(const uint4*)(bp2);
    bf2.q[1] = *(const uint4*)(bp2 + 8);
    bf3.q[0] = *(const uint4*)(bp3);
    bf3.q[1] = *(const uint4*)(bp3 + 8);

    acc0 = __builtin_amdgcn_wmma_f32_16x16x32_bf16(false, af.v, false, bf0.v,
                                                   (short)0, acc0, false, false);
    acc1 = __builtin_amdgcn_wmma_f32_16x16x32_bf16(false, af.v, false, bf1.v,
                                                   (short)0, acc1, false, false);
    acc2 = __builtin_amdgcn_wmma_f32_16x16x32_bf16(false, af.v, false, bf2.v,
                                                   (short)0, acc2, false, false);
    acc3 = __builtin_amdgcn_wmma_f32_16x16x32_bf16(false, af.v, false, bf3.v,
                                                   (short)0, acc3, false, false);
    ap += 32; bp0 += 32; bp1 += 32; bp2 += 32; bp3 += 32;
  }

  // store: VGPR i -> row m0 + half*8 + i, col = n0 + 16*t + r
  const int rowb = m0 + half * 8;
  if (m0 + 16 <= M && n0 + 64 <= N) {          // fast full-tile path
    float* cp = C + (size_t)rowb * ldc + n0 + r;
#pragma unroll
    for (int i = 0; i < 8; ++i) {
      cp[0]  = acc0[i];
      cp[16] = acc1[i];
      cp[32] = acc2[i];
      cp[48] = acc3[i];
      cp += ldc;
    }
  } else {                                     // guarded tail path (N=24 case)
    float* cp = C + (size_t)rowb * ldc;
#pragma unroll
    for (int i = 0; i < 8; ++i) {
      int row = rowb + i;
      if (row < M) {
        if (bcol[0] < N) cp[bcol[0]] = acc0[i];
        if (bcol[1] < N) cp[bcol[1]] = acc1[i];
        if (bcol[2] < N) cp[bcol[2]] = acc2[i];
        if (bcol[3] < N) cp[bcol[3]] = acc3[i];
      }
      cp += ldc;
    }
  }
}

// ------------------------------------------------- conv + SiLU + l2norm -----
__global__ __launch_bounds__(256) void conv_silu_norm_kernel(
    const float* __restrict__ pre, const float* __restrict__ w,
    float* __restrict__ out, float scale) {
  int bl = blockIdx.x;            // b*L + l
  int h  = blockIdx.y;
  int t  = threadIdx.x;
  int l  = bl & (L_ - 1);
  int c  = h * HD_ + t;
  float y = 0.f;
#pragma unroll
  for (int i = 0; i < 4; ++i) {
    int li = l - 3 + i;
    if (li >= 0) y += pre[(size_t)(bl - 3 + i) * KD_ + c] * w[c * 4 + i];
  }
  y = y * sigmoidf_(y);
  __shared__ float red[256];
  red[t] = y * y;
  __syncthreads();
  for (int s = 128; s > 0; s >>= 1) {
    if (t < s) red[t] += red[t + s];
    __syncthreads();
  }
  float rs = rsqrtf(red[0] + 1e-6f) * scale;
  out[(size_t)bl * KD_ + c] = y * rs;
}

// --------------------------------------------------- conv + SiLU (v path) ---
__global__ __launch_bounds__(256) void conv_silu_v_kernel(
    const float* __restrict__ pre, const float* __restrict__ w,
    float* __restrict__ out) {
  size_t idx = (size_t)blockIdx.x * 256 + threadIdx.x;   // < ML*VD
  int c  = (int)(idx % VD_);
  long row = (long)(idx / VD_);
  int l  = (int)(row & (L_ - 1));
  float y = 0.f;
#pragma unroll
  for (int i = 0; i < 4; ++i) {
    int li = l - 3 + i;
    if (li >= 0) y += pre[(size_t)(row - 3 + i) * VD_ + c] * w[c * 4 + i];
  }
  out[idx] = y * sigmoidf_(y);
}

// ------------------------------------------------------ gating scalars ------
__global__ __launch_bounds__(256) void scalars_kernel(
    const float* __restrict__ bpre, const float* __restrict__ apre,
    const float* __restrict__ A_log, const float* __restrict__ dt_bias,
    const int* __restrict__ mod, float* __restrict__ decA,
    float* __restrict__ bmA) {
  int idx = blockIdx.x * 256 + threadIdx.x;   // < E*B*NH*L
  int h = idx % NH_;
  int e = (idx / NH_) % E_;
  int l = (idx / (NH_ * E_)) % L_;
  int b = idx / (NH_ * E_ * L_);
  int ch = e * NH_ + h;
  int bl = b * L_ + l;
  float beta = sigmoidf_(bpre[(size_t)bl * (NH_ * E_) + ch]);
  float t = apre[(size_t)bl * (NH_ * E_) + ch] + dt_bias[ch];
  float sp = (t > 20.f) ? t : log1pf(expf(t));
  float g = -expf(A_log[ch]) * sp;
  int mi = mod[bl];
  float m = (e == 0) ? 1.f : ((e == 1) ? (mi == 0 ? 1.f : 0.f)
                                       : (mi == 1 ? 1.f : 0.f));
  size_t o = ((size_t)(e * B_ + b) * NH_ + h) * L_ + l;
  decA[o] = (m > 0.f) ? expf(g) : 1.f;   // exp(g)*m + (1-m)
  bmA[o]  = beta * m;
}

// ------------------------------------------------------ softmax of w_out ----
__global__ void wout_kernel(const float* __restrict__ ow, float* __restrict__ wout) {
  int h = threadIdx.x;
  if (h >= NH_) return;
  float a0 = ow[h], a1 = ow[NH_ + h], a2 = ow[2 * NH_ + h];
  float mx = fmaxf(a0, fmaxf(a1, a2));
  float e0 = expf(a0 - mx), e1 = expf(a1 - mx), e2 = expf(a2 - mx);
  float s = e0 + e1 + e2;
  wout[h] = e0 / s; wout[NH_ + h] = e1 / s; wout[2 * NH_ + h] = e2 / s;
}

// ------------------------------------------------------------- the scan -----
// grid = (B*NH, DV/128); block = 512 threads (16 waves).
// wave w handles v = chunk*128 + w*8 + (lane&7); lane>>3 = k-quarter (64 rows).
// Per-lane state: S[e][64] in VGPRs for all 3 experts (deterministic expert
// aggregation in-register).  Per-step k/q/v staged to padded LDS.
#define LK(e, kq, j) (lds[(e) * 260 + (kq) * 65 + (j)])
#define LQ(kq, j)    (lds[780 + (kq) * 65 + (j)])
#define LV(i)        (lds[1040 + (i)])
#define LSC(i)       (lds[1168 + (i)])

__global__ __launch_bounds__(512, 1) void scan_kernel(
    const float* __restrict__ ke, const float* __restrict__ qn,
    const float* __restrict__ vc, const float* __restrict__ decA,
    const float* __restrict__ bmA, const float* __restrict__ wout,
    float* __restrict__ obuf) {
  const int b = blockIdx.x / NH_;
  const int h = blockIdx.x % NH_;
  const int chunk = blockIdx.y;          // 0..3, 128 v each
  const int tid = threadIdx.x;
  const int w = tid >> 5;
  const int lane = tid & 31;
  const int kq = (lane >> 3) & 3;
  const int vloc = w * 8 + (lane & 7);   // 0..127

  const float* keB0 = ke + (size_t)0 * ML_ * KD_ + (size_t)(b * L_) * KD_ + h * HD_;
  const float* keB1 = ke + (size_t)1 * ML_ * KD_ + (size_t)(b * L_) * KD_ + h * HD_;
  const float* keB2 = ke + (size_t)2 * ML_ * KD_ + (size_t)(b * L_) * KD_ + h * HD_;
  const float* qB = qn + (size_t)(b * L_) * KD_ + h * HD_;
  const float* vB = vc + (size_t)(b * L_) * VD_ + h * DV_ + chunk * 128;
  const float* decB[3], *bmB[3];
#pragma unroll
  for (int e = 0; e < 3; ++e) {
    size_t off = ((size_t)(e * B_ + b) * NH_ + h) * L_;
    decB[e] = decA + off;
    bmB[e] = bmA + off;
  }
  float* oP = obuf + ((size_t)(b * L_) * NH_ + h) * DV_ + chunk * 128;
  const float w0 = wout[h], w1 = wout[NH_ + h], w2 = wout[2 * NH_ + h];

  __shared__ float lds[1184];

  float S0[64], S1[64], S2[64];
#pragma unroll
  for (int j = 0; j < 64; ++j) { S0[j] = 0.f; S1[j] = 0.f; S2[j] = 0.f; }

  for (int l = 0; l < L_; ++l) {
    // ---- stage k(e=0..2), q, v-chunk, scalars into LDS -------------------
    for (int idx = tid; idx < 1152; idx += 512) {
      const float* g;
      int dst;
      if (idx < 768) {
        int e = idx >> 8, k = idx & 255;
        const float* base = (e == 0) ? keB0 : (e == 1) ? keB1 : keB2;
        g = base + (size_t)l * KD_ + k;
        dst = e * 260 + (k >> 6) * 65 + (k & 63);
      } else if (idx < 1024) {
        int k = idx - 768;
        g = qB + (size_t)l * KD_ + k;
        dst = 780 + (k >> 6) * 65 + (k & 63);
      } else {
        int j = idx - 1024;
        g = vB + (size_t)l * VD_ + j;
        dst = 1040 + j;
      }
      stage_f32(g, &lds[dst]);
    }
    if (tid < 3)      stage_f32(decB[tid] + l, &lds[1168 + tid]);
    else if (tid < 6) stage_f32(bmB[tid - 3] + l, &lds[1171 + tid - 3]);
    stage_wait();
    __syncthreads();

    if (l + 1 < L_) {   // hint next step's rows toward the caches
      __builtin_prefetch(keB0 + (size_t)(l + 1) * KD_, 0, 1);
      __builtin_prefetch(qB + (size_t)(l + 1) * KD_, 0, 1);
      __builtin_prefetch(vB + (size_t)(l + 1) * VD_, 0, 1);
    }

    const float dec0 = LSC(0), dec1 = LSC(1), dec2 = LSC(2);
    const float bm0 = LSC(3), bm1 = LSC(4), bm2 = LSC(5);

    // ---- pred = k . (dec * S)  (dec folded in after the reduction) -------
    float p0 = 0.f, p1 = 0.f, p2 = 0.f;
#pragma unroll
    for (int j = 0; j < 64; ++j) {
      p0 += LK(0, kq, j) * S0[j];
      p1 += LK(1, kq, j) * S1[j];
      p2 += LK(2, kq, j) * S2[j];
    }
    p0 += __shfl_xor(p0, 8, 32);  p0 += __shfl_xor(p0, 16, 32);
    p1 += __shfl_xor(p1, 8, 32);  p1 += __shfl_xor(p1, 16, 32);
    p2 += __shfl_xor(p2, 8, 32);  p2 += __shfl_xor(p2, 16, 32);

    const float vv = LV(vloc);
    const float d0 = (vv - dec0 * p0) * bm0;
    const float d1 = (vv - dec1 * p1) * bm1;
    const float d2 = (vv - dec2 * p2) * bm2;

    // ---- S = S*dec + k (x) delta ;  o_e += q . S --------------------------
    float o0 = 0.f, o1 = 0.f, o2 = 0.f;
#pragma unroll
    for (int j = 0; j < 64; ++j) {
      float qq = LQ(kq, j);
      S0[j] = S0[j] * dec0 + LK(0, kq, j) * d0;  o0 += qq * S0[j];
      S1[j] = S1[j] * dec1 + LK(1, kq, j) * d1;  o1 += qq * S1[j];
      S2[j] = S2[j] * dec2 + LK(2, kq, j) * d2;  o2 += qq * S2[j];
    }
    float oc = w0 * o0 + w1 * o1 + w2 * o2;
    oc += __shfl_xor(oc, 8, 32);
    oc += __shfl_xor(oc, 16, 32);
    if (kq == 0) oP[(size_t)l * NH_ * DV_ + vloc] = oc;
    __syncthreads();   // before next step's staging overwrites LDS
  }
}

// --------------------------------------- RMS norm + swish gate -> og --------
__global__ __launch_bounds__(256) void gate_norm_kernel(
    const float* __restrict__ o, const float* __restrict__ gate,
    const float* __restrict__ nw, bf16* __restrict__ og) {
  int bl = blockIdx.x, h = blockIdx.y, t = threadIdx.x;
  const float* op = o + ((size_t)bl * NH_ + h) * DV_;
  float o0 = op[t], o1 = op[t + 256];
  __shared__ float red[256];
  red[t] = o0 * o0 + o1 * o1;
  __syncthreads();
  for (int s = 128; s > 0; s >>= 1) {
    if (t < s) red[t] += red[t + s];
    __syncthreads();
  }
  float rs = rsqrtf(red[0] / (float)DV_ + 1e-5f);
  const float* gp = gate + (size_t)bl * VD_ + h * DV_;
  float g0 = gp[t], g1 = gp[t + 256];
  bf16* out = og + (size_t)bl * VD_ + h * DV_;
  union { unsigned short u; bf16 b; } r0, r1;
  r0.u = f2bf(o0 * rs * nw[t]       * (g0 * sigmoidf_(g0)));
  r1.u = f2bf(o1 * rs * nw[t + 256] * (g1 * sigmoidf_(g1)));
  out[t]       = r0.b;
  out[t + 256] = r1.b;
}

// ----------------------------------------------------------------- launch ---
extern "C" void kernel_launch(void* const* d_in, const int* in_sizes, int n_in,
                              void* d_out, int out_size, void* d_ws, size_t ws_size,
                              hipStream_t stream) {
  const float* x      = (const float*)d_in[0];
  const int*   mod    = (const int*)d_in[1];
  const float* Wq     = (const float*)d_in[2];
  const float* Wk     = (const float*)d_in[3];
  const float* Wv     = (const float*)d_in[4];
  const float* conv_q = (const float*)d_in[5];
  const float* conv_k = (const float*)d_in[6];
  const float* conv_v = (const float*)d_in[7];
  const float* W_ek   = (const float*)d_in[8];
  const float* Wb     = (const float*)d_in[9];
  const float* Wa     = (const float*)d_in[10];
  const float* A_log  = (const float*)d_in[11];
  const float* dt_b   = (const float*)d_in[12];
  const float* ow     = (const float*)d_in[13];
  const float* Wg     = (const float*)d_in[14];
  const float* onw    = (const float*)d_in[15];
  const float* Wo     = (const float*)d_in[16];
  float* out = (float*)d_out;

  float* ws = (float*)d_ws;
  size_t off = 0;
  auto alloc = [&](size_t n) { float* p = ws + off; off += (n + 63) & ~(size_t)63; return p; };
  auto allocB = [&](size_t n) { return (bf16*)alloc((n + 1) / 2); };  // n bf16 elems

  float* R1   = alloc((size_t)ML_ * VD_);        // transient: q/k/v pre
  float* qn   = alloc((size_t)ML_ * KD_);
  float* kn   = alloc((size_t)ML_ * KD_);
  float* vc   = alloc((size_t)ML_ * VD_);
  float* gate = alloc((size_t)ML_ * VD_);
  float* keb  = alloc((size_t)E_ * ML_ * KD_);
  float* bpre = alloc((size_t)ML_ * NH_ * E_);
  float* apre = alloc((size_t)ML_ * NH_ * E_);
  float* decA = alloc((size_t)E_ * B_ * NH_ * L_);
  float* bmA  = alloc((size_t)E_ * B_ * NH_ * L_);
  float* wout = alloc(64);
  float* obuf = alloc((size_t)ML_ * NH_ * DV_);
  // bf16 operand pool
  bf16* xbf   = allocB((size_t)ML_ * HS_);
  bf16* wqbf  = allocB((size_t)KD_ * HS_);
  bf16* wkbf  = allocB((size_t)KD_ * HS_);
  bf16* wvbf  = allocB((size_t)VD_ * HS_);
  bf16* wgbf  = allocB((size_t)VD_ * HS_);
  bf16* wbbf  = allocB((size_t)NH_ * E_ * HS_);
  bf16* wabf  = allocB((size_t)NH_ * E_ * HS_);
  bf16* wekbf = allocB((size_t)E_ * NH_ * HD_ * HD_);
  bf16* wobf  = allocB((size_t)HS_ * VD_);
  bf16* knbf  = allocB((size_t)ML_ * KD_);
  bf16* ogbf  = allocB((size_t)ML_ * VD_);

  auto cvt = [&](const float* in, bf16* o, long n) {
    cvt_bf16_kernel<<<(unsigned)((n / 4 + 255) / 256), 256, 0, stream>>>(in, o, n);
  };
  auto gemm = [&](const bf16* A, const bf16* Bm, float* C, int M, int N, int K,
                  int lda, int ldb, int ldc, long aSE, long aSH, long bSE,
                  long bSH, long cSE, long cSH, int nh, int batches) {
    dim3 g((N + 127) / 128, (M + 63) / 64, batches);
    wmma_gemm_kernel<<<g, 256, 0, stream>>>(A, Bm, C, M, N, K, lda, ldb, ldc,
                                            aSE, aSH, bSE, bSH, cSE, cSH, nh);
  };

  // one-time bf16 conversions of activations + weights
  cvt(x, xbf, (long)ML_ * HS_);
  cvt(Wq, wqbf, (long)KD_ * HS_);
  cvt(Wk, wkbf, (long)KD_ * HS_);
  cvt(Wv, wvbf, (long)VD_ * HS_);
  cvt(Wg, wgbf, (long)VD_ * HS_);
  cvt(Wb, wbbf, (long)NH_ * E_ * HS_);
  cvt(Wa, wabf, (long)NH_ * E_ * HS_);
  cvt(W_ek, wekbf, (long)E_ * NH_ * HD_ * HD_);
  cvt(Wo, wobf, (long)HS_ * VD_);

  // q path
  gemm(xbf, wqbf, R1, ML_, KD_, HS_, HS_, HS_, KD_, 0, 0, 0, 0, 0, 0, 1, 1);
  conv_silu_norm_kernel<<<dim3(ML_, NH_), 256, 0, stream>>>(R1, conv_q, qn, SCALE_);
  // k path
  gemm(xbf, wkbf, R1, ML_, KD_, HS_, HS_, HS_, KD_, 0, 0, 0, 0, 0, 0, 1, 1);
  conv_silu_norm_kernel<<<dim3(ML_, NH_), 256, 0, stream>>>(R1, conv_k, kn, 1.0f);
  cvt(kn, knbf, (long)ML_ * KD_);
  // v path
  gemm(xbf, wvbf, R1, ML_, VD_, HS_, HS_, HS_, VD_, 0, 0, 0, 0, 0, 0, 1, 1);
  conv_silu_v_kernel<<<(ML_ * (size_t)VD_) / 256, 256, 0, stream>>>(R1, conv_v, vc);
  // gate projection
  gemm(xbf, wgbf, gate, ML_, VD_, HS_, HS_, HS_, VD_, 0, 0, 0, 0, 0, 0, 1, 1);
  // beta / a projections (N = 24, guarded tail path)
  gemm(xbf, wbbf, bpre, ML_, NH_ * E_, HS_, HS_, HS_, NH_ * E_, 0, 0, 0, 0, 0, 0, 1, 1);
  gemm(xbf, wabf, apre, ML_, NH_ * E_, HS_, HS_, HS_, NH_ * E_, 0, 0, 0, 0, 0, 0, 1, 1);
  // per-expert per-head key transform: batched 2048x256x256 GEMMs
  gemm(knbf, wekbf, keb, ML_, HD_, HD_, KD_, HD_, KD_,
       /*aSE*/0, /*aSH*/HD_,
       /*bSE*/(long)NH_ * HD_ * HD_, /*bSH*/(long)HD_ * HD_,
       /*cSE*/(long)ML_ * KD_, /*cSH*/HD_, NH_, E_ * NH_);
  // gating scalars + output-weight softmax
  scalars_kernel<<<(E_ * B_ * NH_ * L_) / 256, 256, 0, stream>>>(
      bpre, apre, A_log, dt_b, mod, decA, bmA);
  wout_kernel<<<1, 32, 0, stream>>>(ow, wout);
  // recurrent scan (state in VGPRs)
  scan_kernel<<<dim3(B_ * NH_, DV_ / 128), 512, 0, stream>>>(
      keb, qn, vc, decA, bmA, wout, obuf);
  // RMS norm + swish gate (writes bf16 directly), then output projection
  gate_norm_kernel<<<dim3(ML_, NH_), 256, 0, stream>>>(obuf, gate, onw, ogbf);
  gemm(ogbf, wobf, out, ML_, HS_, VD_, VD_, VD_, HS_, 0, 0, 0, 0, 0, 0, 1, 1);
}
